// GGN_62156766707825
// MI455X (gfx1250) — matrix operand order repaired and checked by reading, not compiled, not run on hardware
//
#include <hip/hip_runtime.h>
#include <hip/hip_bf16.h>
#include <math.h>

typedef __attribute__((ext_vector_type(16))) __bf16        v16bf;
typedef __attribute__((ext_vector_type(8)))  float         v8f;
typedef __attribute__((ext_vector_type(4)))  unsigned int  u32x4;
typedef __attribute__((ext_vector_type(8)))  unsigned int  u32x8;

#define NNODES  100000
#define NEDGES  1600000
#define CF      128
#define NLAYERS 3
#define TILES   (NNODES / 16)   // 6250, exact

// ---------------------------------------------------------------- helpers

__device__ __forceinline__ unsigned short f2bf(float f) {
  unsigned int u = __builtin_bit_cast(unsigned int, f);
  u += 0x7FFFu + ((u >> 16) & 1u);          // round-to-nearest-even
  return (unsigned short)(u >> 16);
}

__device__ __forceinline__ v16bf make_frag(u32x4 lo, u32x4 hi) {
  u32x8 w = __builtin_shufflevector(lo, hi, 0, 1, 2, 3, 4, 5, 6, 7);
  return __builtin_bit_cast(v16bf, w);
}

// A-fragment (16x32 bf16, M x K) from LDS tile [16 rows][128 k], row-major.
// ISA layout: lanes 0-15 -> K {kb..kb+7},{kb+16..kb+23}; lanes 16-31 -> +8.
__device__ __forceinline__ v16bf load_a_frag(const unsigned short* As,
                                             int l16, int half, int kb) {
  const unsigned short* p = As + l16 * CF + kb + (half ? 8 : 0);
  u32x4 lo = *(const u32x4*)(p);
  u32x4 hi = *(const u32x4*)(p + 16);
  return make_frag(lo, hi);
}

// B-fragment (32x16 bf16, K x N) from global bf16 weights stored [n][k].
// ISA layout: lanes 0-15 -> K {kb..kb+7},{kb+16..kb+23}; lanes 16-31 -> +8.
// ncol already includes this lane's column (l16).
__device__ __forceinline__ v16bf load_b_frag(const unsigned short* Bt,
                                             int ncol, int half, int kb) {
  const unsigned short* p = Bt + (size_t)ncol * CF + kb + half * 8;
  u32x4 lo = *(const u32x4*)(p);
  u32x4 hi = *(const u32x4*)(p + 16);
  return make_frag(lo, hi);
}

__device__ __forceinline__ v8f wmma_bf16(v16bf a, v16bf b, v8f c) {
  return __builtin_amdgcn_wmma_f32_16x16x32_bf16(false, a, false, b,
                                                 (short)0, c, false, false);
}

__device__ __forceinline__ float sigmoidf_(float x) {
  return 1.0f / (1.0f + __expf(-x));
}

// ---------------------------------------------------------------- kernels

// In-degree counts.
__global__ __launch_bounds__(256) void k_degree(const long long* __restrict__ dst,
                                                float* __restrict__ cnt) {
  int e = blockIdx.x * 256 + threadIdx.x;
  if (e >= NEDGES) return;
  atomicAdd(&cnt[(int)dst[e]], 1.0f);
}

__global__ __launch_bounds__(256) void k_inv(const float* __restrict__ cnt,
                                             float* __restrict__ inv) {
  int n = blockIdx.x * 256 + threadIdx.x;
  if (n < NNODES) inv[n] = 1.0f / fmaxf(cnt[n], 1.0f);
}

// Convert weights to bf16 once. wtb[l][n][k] = weight[l][k][n] (transposed),
// wihb/whhb are already [row=3C][k=C] which is exactly the B^T layout we need.
__global__ __launch_bounds__(256) void k_prep(const float* __restrict__ w,
                                              const float* __restrict__ wih,
                                              const float* __restrict__ whh,
                                              unsigned short* __restrict__ wtb,
                                              unsigned short* __restrict__ wihb,
                                              unsigned short* __restrict__ whhb) {
  int t = blockIdx.x * 256 + threadIdx.x;
  const int S = 3 * CF * CF;   // 49152
  if (t < S) {
    int l = t / (CF * CF), rem = t % (CF * CF);
    int n = rem / CF, k = rem % CF;
    wtb[t] = f2bf(w[l * CF * CF + k * CF + n]);
  } else if (t < 2 * S) {
    wihb[t - S] = f2bf(wih[t - S]);
  } else if (t < 3 * S) {
    whhb[t - 2 * S] = f2bf(whh[t - 2 * S]);
  }
}

// m = h @ W   (W given transposed-bf16 as Wt[n][k]).
__global__ __launch_bounds__(128) void k_gemm_m(const float* __restrict__ h,
                                                const unsigned short* __restrict__ Wt,
                                                float* __restrict__ m) {
  __shared__ unsigned short hs[4][16 * CF];
  int wv = threadIdx.x >> 5, lid = threadIdx.x & 31;
  int tile = blockIdx.x * 4 + wv;
  if (tile >= TILES) return;

  unsigned short* Hs = hs[wv];
  const float* hrow = h + (size_t)tile * 16 * CF;
  // Convert this wave's 16x128 h tile to bf16 in LDS (wave-private slice).
  for (int i = 0; i < 16; ++i) {
    int e4 = (i * 32 + lid) * 4;
    int r = e4 >> 7, c = e4 & 127;
    float4 v = *(const float4*)(hrow + r * CF + c);
    unsigned p0 = (unsigned)f2bf(v.x) | ((unsigned)f2bf(v.y) << 16);
    unsigned p1 = (unsigned)f2bf(v.z) | ((unsigned)f2bf(v.w) << 16);
    *(uint2*)(Hs + e4) = make_uint2(p0, p1);
  }

  int half = lid >> 4, l16 = lid & 15;
  for (int ct = 0; ct < 8; ++ct) {
    v8f acc = {};
#pragma unroll
    for (int ks = 0; ks < 4; ++ks) {
      v16bf a = load_a_frag(Hs, l16, half, ks * 32);
      v16bf b = load_b_frag(Wt, ct * 16 + l16, half, ks * 32);
      acc = wmma_bf16(a, b, acc);
    }
    float* mt = m + (size_t)tile * 16 * CF + ct * 16;
#pragma unroll
    for (int v = 0; v < 8; ++v)
      mt[(v + half * 8) * CF + l16] = acc[v];
  }
}

// agg[dst] += m[src]  (each thread: one edge x 4 features).
__global__ __launch_bounds__(256) void k_scatter(const long long* __restrict__ src,
                                                 const long long* __restrict__ dst,
                                                 const float* __restrict__ m,
                                                 float* __restrict__ agg) {
  long long t = (long long)blockIdx.x * 256 + threadIdx.x;
  int e = (int)(t >> 5);
  int lane = (int)(t & 31);
  if (e >= NEDGES) return;
  int s = (int)src[e], d = (int)dst[e];
  float4 v = *(const float4*)(m + (size_t)s * CF + lane * 4);
  float* o = agg + (size_t)d * CF + lane * 4;
  atomicAdd(o + 0, v.x);
  atomicAdd(o + 1, v.y);
  atomicAdd(o + 2, v.z);
  atomicAdd(o + 3, v.w);
}

// Fused GRU cell: 6 WMMA GEMM streams + gates, h updated in place.
__global__ __launch_bounds__(128) void k_gru(const float* __restrict__ agg,
                                             const float* __restrict__ inv,
                                             float* __restrict__ h,
                                             const unsigned short* __restrict__ wih,
                                             const unsigned short* __restrict__ whh,
                                             const float* __restrict__ bih,
                                             const float* __restrict__ bhh) {
  __shared__ unsigned short sm[4][2][16 * CF];
  int wv = threadIdx.x >> 5, lid = threadIdx.x & 31;
  int tile = blockIdx.x * 4 + wv;
  if (tile >= TILES) return;

  unsigned short* As  = sm[wv][0];   // agg / denom, bf16
  unsigned short* Hsb = sm[wv][1];   // h, bf16
  const float* ap = agg + (size_t)tile * 16 * CF;
  float*       hp = h   + (size_t)tile * 16 * CF;

  for (int i = 0; i < 16; ++i) {
    int e4 = (i * 32 + lid) * 4;
    int r = e4 >> 7, c = e4 & 127;
    float s = inv[tile * 16 + r];
    float4 va = *(const float4*)(ap + r * CF + c);
    float4 vh = *(const float4*)(hp + r * CF + c);
    unsigned a0 = (unsigned)f2bf(va.x * s) | ((unsigned)f2bf(va.y * s) << 16);
    unsigned a1 = (unsigned)f2bf(va.z * s) | ((unsigned)f2bf(va.w * s) << 16);
    *(uint2*)(As + e4) = make_uint2(a0, a1);
    unsigned h0 = (unsigned)f2bf(vh.x) | ((unsigned)f2bf(vh.y) << 16);
    unsigned h1 = (unsigned)f2bf(vh.z) | ((unsigned)f2bf(vh.w) << 16);
    *(uint2*)(Hsb + e4) = make_uint2(h0, h1);
  }

  int half = lid >> 4, l16 = lid & 15;
  for (int ct = 0; ct < 8; ++ct) {
    int col = ct * 16 + l16;
    v8f ir = {}, iz = {}, in_ = {}, hr = {}, hz = {}, hn = {};
#pragma unroll
    for (int ks = 0; ks < 4; ++ks) {
      int kb = ks * 32;
      v16bf aA = load_a_frag(As,  l16, half, kb);
      v16bf aH = load_a_frag(Hsb, l16, half, kb);
      ir  = wmma_bf16(aA, load_b_frag(wih, col,          half, kb), ir);
      iz  = wmma_bf16(aA, load_b_frag(wih, CF + col,     half, kb), iz);
      in_ = wmma_bf16(aA, load_b_frag(wih, 2 * CF + col, half, kb), in_);
      hr  = wmma_bf16(aH, load_b_frag(whh, col,          half, kb), hr);
      hz  = wmma_bf16(aH, load_b_frag(whh, CF + col,     half, kb), hz);
      hn  = wmma_bf16(aH, load_b_frag(whh, 2 * CF + col, half, kb), hn);
    }
    float bir = bih[col],          bhr = bhh[col];
    float biz = bih[CF + col],     bhz = bhh[CF + col];
    float bin = bih[2 * CF + col], bhn = bhh[2 * CF + col];
#pragma unroll
    for (int v = 0; v < 8; ++v) {
      int row = v + half * 8;
      float r  = sigmoidf_(ir[v] + bir + hr[v] + bhr);
      float z  = sigmoidf_(iz[v] + biz + hz[v] + bhz);
      float nn = tanhf(in_[v] + bin + r * (hn[v] + bhn));
      float hv = hp[row * CF + col];
      hp[row * CF + col] = (1.0f - z) * nn + z * hv;
    }
  }
}

// out[n] = tanh(mean(h[n,:])) — one wave per node.
__global__ __launch_bounds__(256) void k_out(const float* __restrict__ h,
                                             float* __restrict__ out) {
  int wv = threadIdx.x >> 5, lane = threadIdx.x & 31;
  int node = blockIdx.x * 8 + wv;
  if (node >= NNODES) return;
  float4 v = ((const float4*)(h + (size_t)node * CF))[lane];
  float s = v.x + v.y + v.z + v.w;
#pragma unroll
  for (int off = 16; off; off >>= 1) s += __shfl_xor(s, off, 32);
  if (lane == 0) out[node] = tanhf(s * (1.0f / (float)CF));
}

// ---------------------------------------------------------------- launch

extern "C" void kernel_launch(void* const* d_in, const int* in_sizes, int n_in,
                              void* d_out, int out_size, void* d_ws, size_t ws_size,
                              hipStream_t stream) {
  (void)in_sizes; (void)n_in; (void)out_size; (void)ws_size;
  const float*     x      = (const float*)d_in[0];
  const long long* ei     = (const long long*)d_in[1];
  const float*     weight = (const float*)d_in[2];
  const float*     w_ih   = (const float*)d_in[3];
  const float*     w_hh   = (const float*)d_in[4];
  const float*     b_ih   = (const float*)d_in[5];
  const float*     b_hh   = (const float*)d_in[6];
  const long long* src = ei;
  const long long* dst = ei + NEDGES;

  char* ws = (char*)d_ws;
  size_t off = 0;
  auto alloc = [&](size_t bytes) -> void* {
    off = (off + 511) & ~(size_t)511;
    void* p = ws + off;
    off += bytes;
    return p;
  };
  float* h   = (float*)alloc((size_t)NNODES * CF * 4);
  float* m   = (float*)alloc((size_t)NNODES * CF * 4);
  float* agg = (float*)alloc((size_t)NNODES * CF * 4);
  float* cnt = (float*)alloc((size_t)NNODES * 4);
  float* inv = (float*)alloc((size_t)NNODES * 4);
  unsigned short* wtb  = (unsigned short*)alloc((size_t)3 * CF * CF * 2);
  unsigned short* wihb = (unsigned short*)alloc((size_t)3 * CF * CF * 2);
  unsigned short* whhb = (unsigned short*)alloc((size_t)3 * CF * CF * 2);

  hipMemcpyAsync(h, x, (size_t)NNODES * CF * 4, hipMemcpyDeviceToDevice, stream);
  hipMemsetAsync(cnt, 0, (size_t)NNODES * 4, stream);
  k_degree<<<(NEDGES + 255) / 256, 256, 0, stream>>>(dst, cnt);
  k_inv<<<(NNODES + 255) / 256, 256, 0, stream>>>(cnt, inv);
  k_prep<<<(3 * 3 * CF * CF + 255) / 256, 256, 0, stream>>>(weight, w_ih, w_hh,
                                                            wtb, wihb, whhb);
  for (int l = 0; l < NLAYERS; ++l) {
    k_gemm_m<<<(TILES + 3) / 4, 128, 0, stream>>>(h, wtb + (size_t)l * CF * CF, m);
    hipMemsetAsync(agg, 0, (size_t)NNODES * CF * 4, stream);
    k_scatter<<<(int)(((size_t)NEDGES * 32) / 256), 256, 0, stream>>>(src, dst, m, agg);
    k_gru<<<(TILES + 3) / 4, 128, 0, stream>>>(agg, inv, h, wihb, whhb, b_ih, b_hh);
  }
  k_out<<<NNODES / 8, 256, 0, stream>>>(h, (float*)d_out);
}
